// Detector_31095563223113
// MI455X (gfx1250) — compile-verified, matrix operand order checked by reading, not run on hardware
//
#include <hip/hip_runtime.h>
#include <math.h>

// ---------------- types ----------------
typedef __bf16  bf16x16 __attribute__((ext_vector_type(16)));
typedef float   f32x8   __attribute__((ext_vector_type(8)));
typedef float   f32x4   __attribute__((ext_vector_type(4)));
typedef unsigned int u32x4 __attribute__((ext_vector_type(4)));
typedef unsigned int u32x2 __attribute__((ext_vector_type(2)));

union Frag {
    bf16x16 v;
    u32x4   q[2];
};

// float -> bf16 (round to nearest even), stored as ushort
__device__ __forceinline__ unsigned short f2bf(float f) {
    unsigned int u = __float_as_uint(f);
    u += 0x7fffu + ((u >> 16) & 1u);
    return (unsigned short)(u >> 16);
}

// ---------------- constants ----------------
// img: [32, 3, 512, 512]; patches: M = 32*32*32 = 32768 rows, K = 768
// feat = patches @ w_patch  -> [32768, 768]
// head = feat @ [w_reg | w_obj | pad]  -> [32768, 48] (36 reg + 9 obj + 3 pad)
#define KDIM   768
#define NHEAD  48
#define MBLK   32
#define APITCH 776   // ushort pitch: 776*2B = 1552B = 388 dwords; 388 % 64 = 4 -> conflict-free
#define OPITCH 52    // f32 pitch for head-output tile

// ---------------- prep: Wt[n][k] = bf16(w_patch[k][n]) ----------------
__global__ void prep_wt(const float* __restrict__ w_patch, unsigned short* __restrict__ Wt) {
    int k = blockIdx.x * 256 + threadIdx.x;   // grid.x = 3 -> k in [0,768)
    int n = blockIdx.y;                       // [0,768)
    Wt[(size_t)n * KDIM + k] = f2bf(w_patch[(size_t)k * KDIM + n]);
}

// ---------------- prep: Wh[n][k] = bf16 of packed head weights ----------------
__global__ void prep_wh(const float* __restrict__ w_reg, const float* __restrict__ w_obj,
                        unsigned short* __restrict__ Wh) {
    int k = blockIdx.x * 256 + threadIdx.x;   // grid.x = 3
    int n = blockIdx.y;                       // [0,48)
    float v = 0.0f;
    if (n < 36)      v = w_reg[(size_t)k * 36 + n];
    else if (n < 45) v = w_obj[(size_t)k * 9 + (n - 36)];
    Wh[(size_t)n * KDIM + k] = f2bf(v);
}

// ---------------- fused detector kernel ----------------
__global__ __launch_bounds__(256) void detector_main(
    const float* __restrict__ img,
    const unsigned short* __restrict__ Wt,
    const unsigned short* __restrict__ Wh,
    const float* __restrict__ b_reg,
    const float* __restrict__ b_obj,
    float* __restrict__ out)
{
    __shared__ unsigned short Albd[MBLK * APITCH];   // A tile (bf16), later reused for feat (bf16)
    __shared__ float          Ot[MBLK * OPITCH];     // head outputs (f32)

    const int tid  = threadIdx.x;
    const int lane = tid & 31;
    const int wv   = tid >> 5;          // wave id 0..7
    const int m0   = blockIdx.x * MBLK; // global row base (1024 blocks)

    const int rowA = lane & 15;         // M (A) / N (B) index within fragment
    const int koff = (lane >> 4) * 16;  // K half selector

    // ---- stage A: gather 32 patch rows from img, convert to bf16 in LDS ----
    for (int t = tid; t < MBLK * 192; t += 256) {
        int m = t / 192;
        int k = (t % 192) * 4;                 // 4 consecutive px
        int gm = m0 + m;
        int fx = gm & 31, fy = (gm >> 5) & 31, bi = gm >> 10;
        int c  = k >> 8, py = (k >> 4) & 15, px = k & 15;
        const float* src = img + (((size_t)(bi * 3 + c) * 512 + (fy * 16 + py)) * 512
                                  + fx * 16 + px);
        f32x4 v = *(const f32x4*)src;
        u32x2 pk;
        pk.x = (unsigned int)f2bf(v.x) | ((unsigned int)f2bf(v.y) << 16);
        pk.y = (unsigned int)f2bf(v.z) | ((unsigned int)f2bf(v.w) << 16);
        *(u32x2*)&Albd[m * APITCH + k] = pk;
    }
    __syncthreads();

    // ---- GEMM1: feat[32][768] = A @ w_patch, each wave: 2 M-tiles x 6 N-tiles ----
    const f32x8 fzero = {0.f, 0.f, 0.f, 0.f, 0.f, 0.f, 0.f, 0.f};
    f32x8 acc[2][6];
#pragma unroll
    for (int mt = 0; mt < 2; ++mt)
#pragma unroll
        for (int i = 0; i < 6; ++i) acc[mt][i] = fzero;

    for (int k0 = 0; k0 < KDIM; k0 += 32) {
        Frag a[2];
#pragma unroll
        for (int mt = 0; mt < 2; ++mt) {
            const unsigned short* p = &Albd[(mt * 16 + rowA) * APITCH + k0 + koff];
            a[mt].q[0] = *(const u32x4*)p;
            a[mt].q[1] = *(const u32x4*)(p + 8);
        }
#pragma unroll
        for (int i = 0; i < 6; ++i) {
            int ntg = wv + 8 * i;                       // global N-tile 0..47
            const unsigned short* p = &Wt[(size_t)(ntg * 16 + rowA) * KDIM + k0 + koff];
            Frag b;
            b.q[0] = *(const u32x4*)p;
            b.q[1] = *(const u32x4*)(p + 8);
#pragma unroll
            for (int mt = 0; mt < 2; ++mt)
                acc[mt][i] = __builtin_amdgcn_wmma_f32_16x16x32_bf16(
                    false, a[mt].v, false, b.v, (short)0, acc[mt][i], false, false);
        }
    }

    // ---- spill feat tile back into LDS as bf16 (reuse A buffer) ----
    __syncthreads();
#pragma unroll
    for (int mt = 0; mt < 2; ++mt)
#pragma unroll
        for (int i = 0; i < 6; ++i) {
            int col   = (wv + 8 * i) * 16 + (lane & 15);
            int rbase = mt * 16 + (lane >> 4) * 8;
#pragma unroll
            for (int r = 0; r < 8; ++r)
                Albd[(rbase + r) * APITCH + col] = f2bf(acc[mt][i][r]);
        }
    __syncthreads();

    // ---- GEMM2: head[32][48] = feat @ [w_reg|w_obj], waves 0..5, one tile each ----
    if (wv < 6) {
        int mt = wv & 1, nt = wv >> 1;
        f32x8 acc2 = fzero;
        for (int k0 = 0; k0 < KDIM; k0 += 32) {
            Frag a, b;
            const unsigned short* pa = &Albd[(mt * 16 + rowA) * APITCH + k0 + koff];
            a.q[0] = *(const u32x4*)pa;
            a.q[1] = *(const u32x4*)(pa + 8);
            const unsigned short* pb = &Wh[(size_t)(nt * 16 + rowA) * KDIM + k0 + koff];
            b.q[0] = *(const u32x4*)pb;
            b.q[1] = *(const u32x4*)(pb + 8);
            acc2 = __builtin_amdgcn_wmma_f32_16x16x32_bf16(
                false, a.v, false, b.v, (short)0, acc2, false, false);
        }
        int col   = nt * 16 + (lane & 15);
        int rbase = mt * 16 + (lane >> 4) * 8;
#pragma unroll
        for (int r = 0; r < 8; ++r)
            Ot[(rbase + r) * OPITCH + col] = acc2[r];
    }
    __syncthreads();

    // ---- decode: 32 rows x 9 anchors -> 7-tuples ----
    const float BOXH_c[9] = {2.f, 2.f, 2.f, 4.f, 4.f, 4.f, 8.f, 8.f, 8.f};
    const float BOXW_c[9] = {2.f, 4.f, 8.f, 2.f, 4.f, 8.f, 2.f, 4.f, 8.f};
    for (int t = tid; t < MBLK * 9; t += 256) {
        int m = t / 9, k = t % 9;
        int gm = m0 + m;
        int fx = gm & 31, fy = (gm >> 5) & 31, bi = gm >> 10;
        float r0 = Ot[m * OPITCH + 4 * k + 0] + b_reg[4 * k + 0];
        float r1 = Ot[m * OPITCH + 4 * k + 1] + b_reg[4 * k + 1];
        float r2 = Ot[m * OPITCH + 4 * k + 2] + b_reg[4 * k + 2];
        float r3 = Ot[m * OPITCH + 4 * k + 3] + b_reg[4 * k + 3];
        float lo = Ot[m * OPITCH + 36 + k] + b_obj[k];
        float obj = 1.0f / (1.0f + __expf(-lo));
        float wc = fx * 16.0f + r0;
        float hc = fy * 16.0f + r1;
        float wa = wc + BOXW_c[k] * r2;
        float ha = hc + BOXH_c[k] * r3;
        float* o = out + (size_t)(gm * 9 + k) * 7;
        o[0] = wc; o[1] = hc; o[2] = wa; o[3] = ha;
        o[4] = (float)bi; o[5] = obj; o[6] = (float)k;
    }
}

// ---------------- launch ----------------
extern "C" void kernel_launch(void* const* d_in, const int* in_sizes, int n_in,
                              void* d_out, int out_size, void* d_ws, size_t ws_size,
                              hipStream_t stream) {
    const float* img     = (const float*)d_in[0];
    const float* w_patch = (const float*)d_in[1];
    const float* w_reg   = (const float*)d_in[2];
    const float* b_reg   = (const float*)d_in[3];
    const float* w_obj   = (const float*)d_in[4];
    const float* b_obj   = (const float*)d_in[5];

    unsigned short* Wt = (unsigned short*)d_ws;               // 768*768 bf16 = 1.18 MB
    unsigned short* Wh = Wt + (size_t)KDIM * KDIM;            // 48*768  bf16 = 74 KB

    prep_wt<<<dim3(3, 768), 256, 0, stream>>>(w_patch, Wt);
    prep_wh<<<dim3(3, 48), 256, 0, stream>>>(w_reg, w_obj, Wh);
    detector_main<<<1024, 256, 0, stream>>>(img, Wt, Wh, b_reg, b_obj, (float*)d_out);
}
